// ImageTransformerLayer_36618891165999
// MI455X (gfx1250) — compile-verified, matrix operand order checked by reading, not run on hardware
//
#include <hip/hip_runtime.h>
#include <hip/hip_bf16.h>

// ---------------- problem constants ----------------
#define BB   4
#define DD   8
#define FF   256          // H*W
#define CC   512
#define RR   64           // reduced channels
#define DF   2048         // D*F
#define MROWS 8192        // B*D*F
#define EPSV 1e-3f

typedef __bf16 bf16;
typedef __attribute__((ext_vector_type(16))) __bf16 v16bf;
typedef __attribute__((ext_vector_type(8)))  __bf16 v8bf;
typedef __attribute__((ext_vector_type(8)))  float   v8f;

#define WMMA_BF16(a, b, c) \
  __builtin_amdgcn_wmma_f32_16x16x32_bf16(false, (a), false, (b), (short)0, (c), false, false)

// Fragment load from a K-contiguous (row-major [rows, K]) bf16 matrix.
// Works for A (row = M = lane%16) and for B when B is stored transposed
// [N, K] (row = N = lane%16).  half h -> K = k0 + 16*(h/8) + 8*hi + (h%8)
// => two contiguous 16-byte loads per lane.
__device__ inline v16bf load_frag_k(const bf16* __restrict__ base, int ld,
                                    int row, int k0, int hi) {
  const bf16* p = base + (size_t)row * ld + k0 + 8 * hi;
  v8bf lo = *reinterpret_cast<const v8bf*>(p);
  v8bf hh = *reinterpret_cast<const v8bf*>(p + 16);
  v16bf r;
  #pragma unroll
  for (int i = 0; i < 8; ++i) { r[i] = lo[i]; r[i + 8] = hh[i]; }
  return r;
}

// ---------------- f32 -> bf16 convert (plain) ----------------
__global__ void itl_cvt_bf16(const float* __restrict__ in, bf16* __restrict__ out, int n) {
  int i = blockIdx.x * blockDim.x + threadIdx.x;
  for (; i < n; i += gridDim.x * blockDim.x) out[i] = (bf16)in[i];
}

// ---------------- f32 [K,N] -> bf16 [N,K] transpose-convert (weights) ------------
__global__ void itl_cvt_bf16_T(const float* __restrict__ in, bf16* __restrict__ out,
                               int K, int N) {
  int i = blockIdx.x * blockDim.x + threadIdx.x;
  int total = K * N;
  for (; i < total; i += gridDim.x * blockDim.x) {
    int k = i / N, n = i - k * N;
    out[(size_t)n * K + k] = (bf16)in[i];
  }
}

// ---------------- WMMA GEMM: out = [opt relu](A@B^T_stored + bias) [+ resid] -----
// A: [M,K] bf16 row-major.  BT: [N,K] bf16 row-major (i.e. B transposed).
// One wave computes a 32(M) x 64(N) tile -> 8 independent WMMA chains.
// Optional outputs: f32 row-major, bf16 row-major, bf16 F-slab-transposed
// (outBT[slab= r/FF][col][r%FF], used to produce V^T per (b,d)).
__global__ void itl_gemm_bf16(const bf16* __restrict__ A, const bf16* __restrict__ BT,
                              const float* __restrict__ bias,
                              const float* __restrict__ resid,
                              float* __restrict__ outF, bf16* __restrict__ outB,
                              bf16* __restrict__ outBT,
                              int M, int N, int K, int doRelu) {
  int wave = (blockIdx.x * blockDim.x + threadIdx.x) >> 5;
  int lane = threadIdx.x & 31;
  int ntiles = N >> 6;                 // 64-wide N chunks
  int mt = wave / ntiles, nb = wave - mt * ntiles;
  if (mt * 32 >= M) return;
  int hi = lane >> 4, ln = lane & 15;
  int row0 = mt * 32 + ln;
  int row1 = row0 + 16;

  v8f acc[2][4];
  #pragma unroll
  for (int m = 0; m < 2; ++m)
    #pragma unroll
    for (int t = 0; t < 4; ++t) acc[m][t] = v8f{};

  for (int k0 = 0; k0 < K; k0 += 32) {
    if (k0 + 64 < K) {
      __builtin_prefetch(A + (size_t)row0 * K + k0 + 64, 0, 0);
      __builtin_prefetch(A + (size_t)row1 * K + k0 + 64, 0, 0);
    }
    v16bf a0 = load_frag_k(A, K, row0, k0, hi);
    v16bf a1 = load_frag_k(A, K, row1, k0, hi);
    #pragma unroll
    for (int t = 0; t < 4; ++t) {
      int n = (nb << 6) + t * 16 + ln;
      v16bf b = load_frag_k(BT, K, n, k0, hi);
      acc[0][t] = WMMA_BF16(a0, b, acc[0][t]);
      acc[1][t] = WMMA_BF16(a1, b, acc[1][t]);
    }
  }

  #pragma unroll
  for (int m = 0; m < 2; ++m) {
    #pragma unroll
    for (int t = 0; t < 4; ++t) {
      int col = (nb << 6) + t * 16 + ln;
      float bc = bias ? bias[col] : 0.0f;
      #pragma unroll
      for (int v = 0; v < 8; ++v) {
        int r = mt * 32 + m * 16 + v + 8 * hi;
        float val = acc[m][t][v] + bc;
        if (doRelu) val = fmaxf(val, 0.0f);
        size_t idx = (size_t)r * N + col;
        if (resid) val += resid[idx];
        if (outF)  outF[idx] = val;
        if (outB)  outB[idx] = (bf16)val;
        if (outBT) {
          int slab = r / FF, within = r - slab * FF;
          outBT[(size_t)slab * N * FF + (size_t)col * FF + within] = (bf16)val;
        }
      }
    }
  }
}

// ---------------- softmax stats: row max + 1/Z over all 2048 keys ----------------
// grid: B * (DF/16) blocks of 256 threads (8 waves).
__global__ void itl_attn_stats(const bf16* __restrict__ Qb, const bf16* __restrict__ Kb,
                               float* __restrict__ mOut, float* __restrict__ rZOut) {
  int blk = blockIdx.x;
  int b = blk / (DF / 16);
  int qt = blk - b * (DF / 16);
  int wave = threadIdx.x >> 5, lane = threadIdx.x & 31;
  int hi = lane >> 4, ln = lane & 15;
  const bf16* Qbat = Qb + (size_t)b * DF * RR;
  const bf16* Kbat = Kb + (size_t)b * DF * RR;
  int qrow = qt * 16 + ln;
  v16bf a0 = load_frag_k(Qbat, RR, qrow, 0, hi);
  v16bf a1 = load_frag_k(Qbat, RR, qrow, 32, hi);

  float m8[8], s8[8];
  #pragma unroll
  for (int v = 0; v < 8; ++v) { m8[v] = -3.0e30f; s8[v] = 0.0f; }

  for (int kt = wave; kt < DF / 16; kt += 8) {
    int krow = kt * 16 + ln;
    v16bf b0 = load_frag_k(Kbat, RR, krow, 0, hi);
    v16bf b1 = load_frag_k(Kbat, RR, krow, 32, hi);
    v8f e = v8f{};
    e = WMMA_BF16(a0, b0, e);
    e = WMMA_BF16(a1, b1, e);
    #pragma unroll
    for (int v = 0; v < 8; ++v) {
      float x = e[v];
      float tm = x;                           // row-wise max across 16 lanes
      #pragma unroll
      for (int msk = 1; msk < 16; msk <<= 1) tm = fmaxf(tm, __shfl_xor(tm, msk, 32));
      float nm = fmaxf(m8[v], tm);
      float p = __expf(x - nm);
      #pragma unroll
      for (int msk = 1; msk < 16; msk <<= 1) p += __shfl_xor(p, msk, 32);
      s8[v] = s8[v] * __expf(m8[v] - nm) + p;
      m8[v] = nm;
    }
  }

  __shared__ float mS[8][16], sS[8][16];
  if (ln == 0) {                               // lane 0 -> rows 0..7, lane 16 -> rows 8..15
    #pragma unroll
    for (int v = 0; v < 8; ++v) { mS[wave][hi * 8 + v] = m8[v]; sS[wave][hi * 8 + v] = s8[v]; }
  }
  __syncthreads();
  if (threadIdx.x < 16) {
    int r = threadIdx.x;
    float M = -3.0e30f;
    for (int w = 0; w < 8; ++w) M = fmaxf(M, mS[w][r]);
    float Z = 0.0f;
    for (int w = 0; w < 8; ++w) Z += sS[w][r] * __expf(mS[w][r] - M);
    size_t o = (size_t)b * DF + qt * 16 + r;
    mOut[o] = M;
    rZOut[o] = 1.0f / Z;
  }
}

// ---------------- attention apply: a2 = sum_g softmax, attnpre = a2 @ V + x ----------
// Vt: per (b,d) slab transposed V, layout [B*D][C][F] bf16 (K-contiguous for WMMA B).
// grid: B*D*(F/16) blocks of 256 threads (8 waves).
__global__ void itl_attn_apply(const bf16* __restrict__ Qb, const bf16* __restrict__ Kb,
                               const bf16* __restrict__ Vt,
                               const float* __restrict__ mRow, const float* __restrict__ rZ,
                               const float* __restrict__ x, float* __restrict__ attnpre) {
  int blk = blockIdx.x;
  int ft = blk % (FF / 16);
  int bd = blk / (FF / 16);
  int d = bd % DD, b = bd / DD;
  int wave = threadIdx.x >> 5, lane = threadIdx.x & 31;
  int hi = lane >> 4, ln = lane & 15;

  __shared__ __align__(16) bf16 a2s[16][FF];   // 16 queries x 256 keys (g-summed), 8KB

  const bf16* Qbat = Qb + (size_t)b * DF * RR;
  const bf16* Kbat = Kb + (size_t)b * DF * RR;
  int q0 = d * FF + ft * 16;                    // first query row within batch

  v16bf a0 = load_frag_k(Qbat, RR, q0 + ln, 0, hi);
  v16bf a1 = load_frag_k(Qbat, RR, q0 + ln, 32, hi);

  float mv[8], rz[8];
  #pragma unroll
  for (int v = 0; v < 8; ++v) {
    size_t qr = (size_t)b * DF + q0 + v + 8 * hi;
    mv[v] = mRow[qr];
    rz[v] = rZ[qr];
  }

  // stage 1: each wave builds 2 of the 16 j-tiles of a2, summing over key depth g
  for (int jt = wave * 2; jt < wave * 2 + 2; ++jt) {
    float a2f[8] = {0, 0, 0, 0, 0, 0, 0, 0};
    for (int g = 0; g < DD; ++g) {
      int krow = g * FF + jt * 16 + ln;
      v16bf b0 = load_frag_k(Kbat, RR, krow, 0, hi);
      v16bf b1 = load_frag_k(Kbat, RR, krow, 32, hi);
      v8f e = v8f{};
      e = WMMA_BF16(a0, b0, e);
      e = WMMA_BF16(a1, b1, e);
      #pragma unroll
      for (int v = 0; v < 8; ++v) a2f[v] += __expf(e[v] - mv[v]) * rz[v];
    }
    #pragma unroll
    for (int v = 0; v < 8; ++v) a2s[v + 8 * hi][jt * 16 + ln] = (bf16)a2f[v];
  }
  __syncthreads();

  // stage 2: attn[16, c0:c0+64] = a2[16,256] @ V[b,d][256, c0:c0+64]
  // B fragments come from Vt (K=j contiguous) -> 2 x b128 per fragment.
  int c0 = wave * 64;
  v8f acc[4] = {v8f{}, v8f{}, v8f{}, v8f{}};
  const bf16* VtSlab = Vt + ((size_t)(b * DD + d) * CC) * FF;
  for (int j0 = 0; j0 < FF; j0 += 32) {
    v16bf a = load_frag_k(&a2s[0][0], FF, ln, j0, hi);
    #pragma unroll
    for (int t = 0; t < 4; ++t) {
      int c = c0 + t * 16 + ln;
      v16bf bb = load_frag_k(VtSlab, FF, c, j0, hi);
      acc[t] = WMMA_BF16(a, bb, acc[t]);
    }
  }

  const float* xrow0 = x + ((size_t)(b * DD + d) * FF) * CC;
  float* orow0 = attnpre + ((size_t)(b * DD + d) * FF) * CC;
  #pragma unroll
  for (int t = 0; t < 4; ++t) {
    int c = c0 + t * 16 + ln;
    #pragma unroll
    for (int v = 0; v < 8; ++v) {
      size_t idx = (size_t)(ft * 16 + v + 8 * hi) * CC + c;
      orow0[idx] = acc[t][v] + xrow0[idx];
    }
  }
}

// ---------------- LayerNorm over C=512: one block (256 thr) per row ----------------
__global__ void itl_ln512(const float* __restrict__ in, const float* __restrict__ gamma,
                          const float* __restrict__ beta, float* __restrict__ outF,
                          bf16* __restrict__ outB) {
  const int CT = CC;
  int row = blockIdx.x, t = threadIdx.x;
  const float* p = in + (size_t)row * CT;
  float v0 = p[t], v1 = p[t + 256];
  __shared__ float red[256];
  red[t] = v0 + v1;
  __syncthreads();
  for (int st = 128; st > 0; st >>= 1) {
    if (t < st) red[t] += red[t + st];
    __syncthreads();
  }
  float mean = red[0] * (1.0f / CT);
  __syncthreads();
  float d0 = v0 - mean, d1 = v1 - mean;
  red[t] = d0 * d0 + d1 * d1;
  __syncthreads();
  for (int st = 128; st > 0; st >>= 1) {
    if (t < st) red[t] += red[t + st];
    __syncthreads();
  }
  float rs = rsqrtf(red[0] * (1.0f / CT) + EPSV);
  float o0 = d0 * rs * gamma[t] + beta[t];
  float o1 = d1 * rs * gamma[t + 256] + beta[t + 256];
  float* q = outF + (size_t)row * CT;
  q[t] = o0;
  q[t + 256] = o1;
  if (outB) {
    outB[(size_t)row * CT + t] = (bf16)o0;
    outB[(size_t)row * CT + t + 256] = (bf16)o1;
  }
}

// ---------------- host-side orchestration ----------------
static inline size_t alignUp(size_t v, size_t a) { return (v + a - 1) & ~(a - 1); }

extern "C" void kernel_launch(void* const* d_in, const int* in_sizes, int n_in,
                              void* d_out, int out_size, void* d_ws, size_t ws_size,
                              hipStream_t stream) {
  (void)in_sizes; (void)n_in; (void)out_size; (void)ws_size;
  const float* x  = (const float*)d_in[0];
  const float* wq = (const float*)d_in[1];
  const float* bq = (const float*)d_in[2];
  const float* wk = (const float*)d_in[3];
  const float* bk = (const float*)d_in[4];
  const float* wv = (const float*)d_in[5];
  const float* bv = (const float*)d_in[6];
  const float* wm = (const float*)d_in[7];
  const float* bm = (const float*)d_in[8];
  const float* g1 = (const float*)d_in[9];
  const float* b1 = (const float*)d_in[10];
  const float* g2 = (const float*)d_in[11];
  const float* b2 = (const float*)d_in[12];
  float* out = (float*)d_out;

  char* ws = (char*)d_ws;
  size_t off = 0;
  auto carve = [&](size_t bytes) { size_t o = off; off = alignUp(off + bytes, 256); return o; };

  const size_t nX = (size_t)MROWS * CC;       // 4,194,304
  bf16*  xb      = (bf16*)(ws + carve(nX * 2));
  bf16*  wqT     = (bf16*)(ws + carve((size_t)CC * RR * 2));   // [R,C]
  bf16*  wkT     = (bf16*)(ws + carve((size_t)CC * RR * 2));   // [R,C]
  bf16*  wvT     = (bf16*)(ws + carve((size_t)CC * CC * 2));   // [C,C] transposed
  bf16*  wmT     = (bf16*)(ws + carve((size_t)CC * CC * 2));   // [C,C] transposed
  bf16*  Qb      = (bf16*)(ws + carve((size_t)MROWS * RR * 2));
  bf16*  Kb      = (bf16*)(ws + carve((size_t)MROWS * RR * 2));
  bf16*  Vt      = (bf16*)(ws + carve(nX * 2));                // [B*D][C][F]
  float* mRow    = (float*)(ws + carve((size_t)MROWS * 4));
  float* rZ      = (float*)(ws + carve((size_t)MROWS * 4));
  float* attnpre = (float*)(ws + carve(nX * 4));
  float* y1      = (float*)(ws + carve(nX * 4));
  bf16*  y1b     = (bf16*)(ws + carve(nX * 2));
  float* y2pre   = (float*)(ws + carve(nX * 4));

  // 1) convert x (plain) and weights (transposed) to bf16
  itl_cvt_bf16  <<<4096, 256, 0, stream>>>(x,  xb,  (int)nX);
  itl_cvt_bf16_T<<<64,   256, 0, stream>>>(wq, wqT, CC, RR);
  itl_cvt_bf16_T<<<64,   256, 0, stream>>>(wk, wkT, CC, RR);
  itl_cvt_bf16_T<<<256,  256, 0, stream>>>(wv, wvT, CC, CC);
  itl_cvt_bf16_T<<<256,  256, 0, stream>>>(wm, wmT, CC, CC);

  // 2) projections (WMMA GEMMs). waves = (M/32)*(N/64); 8 waves/block.
  {
    int waves = (MROWS / 32) * (RR / 64);      // 256
    itl_gemm_bf16<<<(waves + 7) / 8, 256, 0, stream>>>(
        xb, wqT, bq, nullptr, nullptr, Qb, nullptr, MROWS, RR, CC, 0);
    itl_gemm_bf16<<<(waves + 7) / 8, 256, 0, stream>>>(
        xb, wkT, bk, nullptr, nullptr, Kb, nullptr, MROWS, RR, CC, 0);
  }
  {
    int waves = (MROWS / 32) * (CC / 64);      // 2048
    itl_gemm_bf16<<<(waves + 7) / 8, 256, 0, stream>>>(
        xb, wvT, bv, nullptr, nullptr, nullptr, Vt, MROWS, CC, CC, 0);
  }

  // 3) softmax stats over all 2048 keys per query row
  itl_attn_stats<<<BB * (DF / 16), 256, 0, stream>>>(Qb, Kb, mRow, rZ);

  // 4) attention apply + residual
  itl_attn_apply<<<BB * DD * (FF / 16), 256, 0, stream>>>(Qb, Kb, Vt, mRow, rZ, x, attnpre);

  // 5) LN1 -> y1 (f32 + bf16)
  itl_ln512<<<MROWS, 256, 0, stream>>>(attnpre, g1, b1, y1, y1b);

  // 6) MLP: y2pre = y1 + relu(y1 @ wm + bm)
  {
    int waves = (MROWS / 32) * (CC / 64);
    itl_gemm_bf16<<<(waves + 7) / 8, 256, 0, stream>>>(
        y1b, wmT, bm, y1, y2pre, nullptr, nullptr, MROWS, CC, CC, 1);
  }

  // 7) LN2 -> out
  itl_ln512<<<MROWS, 256, 0, stream>>>(y2pre, g2, b2, out, nullptr);
}
